// TextMoE_44873818308996
// MI455X (gfx1250) — compile-verified, matrix operand order checked by reading, not run on hardware
//
#include <hip/hip_runtime.h>
#include <math.h>

// ---------------- problem constants ----------------
#define B_  16
#define S_  256
#define D_  512
#define H_  8
#define HS_ 64
#define O_  1024
#define E_  8
#define FF_ 4096
#define T_  (B_*S_)

// ---------------- types / helpers ----------------
typedef __bf16 bf16x16 __attribute__((ext_vector_type(16)));
typedef float  f32x8   __attribute__((ext_vector_type(8)));
typedef int    v4i     __attribute__((ext_vector_type(4)));

__device__ __forceinline__ unsigned short f2bf(float f) {
  unsigned int u = __float_as_uint(f);
  u += 0x7fffu + ((u >> 16) & 1u);            // round-to-nearest-even
  return (unsigned short)(u >> 16);
}
__device__ __forceinline__ float bf2f(unsigned short h) {
  return __uint_as_float(((unsigned int)h) << 16);
}

union Frag { bf16x16 v; uint4 q[2]; };

// ---- CDNA5 async memory->LDS copy (guarded: falls back to sync path) ----
#if defined(__AMDGCN__) && __has_builtin(__builtin_amdgcn_global_load_async_to_lds_b128) && __has_builtin(__builtin_amdgcn_s_wait_asynccnt)
#define USE_ASYNC_COPY 1
#else
#define USE_ASYNC_COPY 0
#endif

#if USE_ASYNC_COPY
#define AS_GLOBAL __attribute__((address_space(1)))
#define AS_LDS    __attribute__((address_space(3)))
__device__ __forceinline__ void async_copy16(const unsigned short* g, unsigned short* l) {
  __builtin_amdgcn_global_load_async_to_lds_b128(
      (AS_GLOBAL v4i*)(g), (AS_LDS v4i*)(l), 0, 0);
}
#endif

// ---------------- generic WMMA GEMM ----------------
// C[M,N] = epilog( A[M,K](bf16) * B[K,N] + bias )
// flags: 1=relu  2=B is fp32  4=B pre-transposed (stored [N][K])  8=store bf16
//        16=accumulate Cf += rowScale[m]*(acc+bias)
struct GemmParams {
  const unsigned short* A;
  const void*  B;
  const float* bias;
  const float* addRes;     // residual (fp32), ld = resLd
  const float* rowScale;   // per-row scale (gate column), stride rsStride
  float*          Cf;
  unsigned short* Cb;
  int M, N, Kd;
  int lda, ldb, ldc;
  long long sAb, sAh, sBb, sBh, sCb, sCh;  // batch strides (elements)
  int Hn;                                   // z -> (b = z/Hn, h = z%Hn)
  int flags;
  int resLd;
  int rsStride;
};

#define BM 128
#define BN 128
#define BK 32

__global__ __launch_bounds__(256) void gemm_bf16_wmma(GemmParams p) {
  __shared__ unsigned short sA[2][BM][BK];   // row-major [m][k], double-buffered
  __shared__ unsigned short sB[2][BN][BK];   // transposed [n][k], double-buffered

  const int tid  = threadIdx.x;
  const int lane = tid & 31;
  const int wave = tid >> 5;
  const int z  = blockIdx.z;
  const int bz = z / p.Hn;
  const int hz = z % p.Hn;

  const int bm = blockIdx.x * BM;
  const int bn = blockIdx.y * BN;

  const bool relu   = (p.flags & 1)  != 0;
  const bool bIsF32 = (p.flags & 2)  != 0;
  const bool bT     = (p.flags & 4)  != 0;
  const bool st16   = (p.flags & 8)  != 0;
  const bool accum  = (p.flags & 16) != 0;

  const unsigned short* A = p.A + bz * p.sAb + hz * p.sAh;
  const unsigned short* Bh = bIsF32 ? nullptr
      : ((const unsigned short*)p.B + bz * p.sBb + hz * p.sBh);
  const float* Bf = bIsF32 ? ((const float*)p.B + bz * p.sBb + hz * p.sBh) : nullptr;

  // wave tiling: 2 (rows) x 4 (cols) waves; each wave -> 64x32 = 4x2 WMMA tiles
  const int wM = (wave & 1) * 64;
  const int wN = (wave >> 1) * 32;

  f32x8 acc[4][2];
#pragma unroll
  for (int i = 0; i < 4; ++i)
#pragma unroll
    for (int j = 0; j < 2; ++j) acc[i][j] = (f32x8)0.0f;

  // fill-index precompute: each thread moves 16 halfs (32 B)
  const int ar = tid >> 1, ac = (tid & 1) << 4;        // A / B-transposed modes
  const int bn_n = tid >> 1, bk0 = (tid & 1) << 4;     // B k-major modes (own a column)

  const bool fullA = (bm + BM <= p.M);                 // uniform per block
  const bool fullB = (bn + BN <= p.N);

  const int hsel = (lane >> 4) ? 8 : 0;
  const int l16  = lane & 15;

  auto fillA = [&](int buf, int kb) {
    const int gm = bm + ar;
    const unsigned short* src = A + (long long)gm * p.lda + kb + ac;
#if USE_ASYNC_COPY
    if (fullA) {
      async_copy16(src,     &sA[buf][ar][ac]);
      async_copy16(src + 8, &sA[buf][ar][ac + 8]);
      return;
    }
#endif
    uint4 v0 = make_uint4(0, 0, 0, 0), v1 = v0;
    if (gm < p.M) {
      v0 = ((const uint4*)src)[0];
      v1 = ((const uint4*)src)[1];
      if (kb + BK < p.Kd) __builtin_prefetch((const void*)(src + BK), 0, 1);
    }
    *(uint4*)&sA[buf][ar][ac]     = v0;
    *(uint4*)&sA[buf][ar][ac + 8] = v1;
  };

  auto fillB = [&](int buf, int kb) {
    if (bT) {                       // B given as [N][K] bf16: straight copy
      const int gn = bn + ar;
      const unsigned short* src = Bh + (long long)gn * p.ldb + kb + ac;
#if USE_ASYNC_COPY
      if (fullB) {
        async_copy16(src,     &sB[buf][ar][ac]);
        async_copy16(src + 8, &sB[buf][ar][ac + 8]);
        return;
      }
#endif
      uint4 v0 = make_uint4(0, 0, 0, 0), v1 = v0;
      if (gn < p.N) {
        v0 = ((const uint4*)src)[0];
        v1 = ((const uint4*)src)[1];
      }
      *(uint4*)&sB[buf][ar][ac]     = v0;
      *(uint4*)&sB[buf][ar][ac + 8] = v1;
    } else if (bIsF32) {            // B [K][N] fp32: convert + transpose, vector DS stores
      const int gn = bn + bn_n;
      unsigned int packed[8];
      const float* src = Bf + (long long)(kb + bk0) * p.ldb + gn;
#pragma unroll
      for (int j = 0; j < 8; ++j) {
        float v0 = 0.0f, v1 = 0.0f;
        if (gn < p.N) {
          v0 = src[(long long)(2 * j) * p.ldb];
          v1 = src[(long long)(2 * j + 1) * p.ldb];
        }
        packed[j] = (unsigned int)f2bf(v0) | ((unsigned int)f2bf(v1) << 16);
      }
      if (gn < p.N && kb + BK < p.Kd)
        __builtin_prefetch((const void*)(src + (long long)BK * p.ldb), 0, 1);
      *(uint4*)&sB[buf][bn_n][bk0]     = *(uint4*)&packed[0];
      *(uint4*)&sB[buf][bn_n][bk0 + 8] = *(uint4*)&packed[4];
    } else {                        // B [K][N] bf16: transpose, vector DS stores
      const int gn = bn + bn_n;
      unsigned int packed[8];
      const unsigned short* src = Bh + (long long)(kb + bk0) * p.ldb + gn;
#pragma unroll
      for (int j = 0; j < 8; ++j) {
        unsigned short v0 = 0, v1 = 0;
        if (gn < p.N) {
          v0 = src[(long long)(2 * j) * p.ldb];
          v1 = src[(long long)(2 * j + 1) * p.ldb];
        }
        packed[j] = (unsigned int)v0 | ((unsigned int)v1 << 16);
      }
      *(uint4*)&sB[buf][bn_n][bk0]     = *(uint4*)&packed[0];
      *(uint4*)&sB[buf][bn_n][bk0 + 8] = *(uint4*)&packed[4];
    }
  };

  auto wait_fills = [&]() {
#if USE_ASYNC_COPY
    __builtin_amdgcn_s_wait_asynccnt(0);
#endif
  };

  // ---- software-pipelined main loop (double-buffered LDS) ----
  fillA(0, 0);
  fillB(0, 0);
  wait_fills();
  __syncthreads();

  const int nk = p.Kd / BK;
  for (int it = 0; it < nk; ++it) {
    const int buf = it & 1;
    if (it + 1 < nk) {                       // stage next tile into other buffer
      fillA(buf ^ 1, (it + 1) * BK);
      fillB(buf ^ 1, (it + 1) * BK);
    }
    // fragments + 8 WMMAs from current buffer
    Frag af[4], bfr[2];
#pragma unroll
    for (int i = 0; i < 4; ++i) {
      const int r = wM + i * 16 + l16;
      af[i].q[0] = *(const uint4*)&sA[buf][r][hsel];
      af[i].q[1] = *(const uint4*)&sA[buf][r][16 + hsel];
    }
#pragma unroll
    for (int j = 0; j < 2; ++j) {
      const int c = wN + j * 16 + l16;
      bfr[j].q[0] = *(const uint4*)&sB[buf][c][hsel];
      bfr[j].q[1] = *(const uint4*)&sB[buf][c][16 + hsel];
    }
#pragma unroll
    for (int i = 0; i < 4; ++i)
#pragma unroll
      for (int j = 0; j < 2; ++j)
        acc[i][j] = __builtin_amdgcn_wmma_f32_16x16x32_bf16(
            false, af[i].v, false, bfr[j].v, (short)0, acc[i][j], false, false);

    wait_fills();
    __syncthreads();
  }

  // ---- epilogue ----
  float*          Cf = p.Cf ? p.Cf + bz * p.sCb + hz * p.sCh : nullptr;
  unsigned short* Cb = p.Cb ? p.Cb + bz * p.sCb + hz * p.sCh : nullptr;

#pragma unroll
  for (int i = 0; i < 4; ++i) {
#pragma unroll
    for (int j = 0; j < 2; ++j) {
      const int col = bn + wN + j * 16 + l16;
      if (col >= p.N) continue;
      const float bias = p.bias ? p.bias[col] : 0.0f;
#pragma unroll
      for (int r = 0; r < 8; ++r) {
        const int m = bm + wM + i * 16 + hsel + r;
        if (m >= p.M) continue;
        float v = acc[i][j][r] + bias;
        if (relu) v = fmaxf(v, 0.0f);
        if (p.addRes) v += p.addRes[(long long)m * p.resLd + col];
        const long long off = (long long)m * p.ldc + col;
        if (accum)      Cf[off] += p.rowScale[(long long)m * p.rsStride] * v;
        else if (st16)  Cb[off] = f2bf(v);
        else            Cf[off] = v;
      }
    }
  }
}

// ---------------- embed + positional ----------------
__global__ __launch_bounds__(256) void embed_kernel(const int* __restrict__ ids,
                                                    const float* __restrict__ emb,
                                                    const float* __restrict__ pe,
                                                    float* __restrict__ x0) {
  const long long i = (long long)blockIdx.x * 256 + threadIdx.x;
  if (i >= (long long)T_ * D_) return;
  const long long t = i / D_;
  const int d = (int)(i % D_);
  const int s = (int)(t % S_);
  const int id = ids[t];
  x0[i] = emb[(long long)id * D_ + d] + pe[(long long)s * D_ + d];
}

// ---------------- layernorm (fp32 in -> bf16 out) ----------------
__global__ __launch_bounds__(256) void ln_kernel(const float* __restrict__ x,
                                                 const float* __restrict__ g,
                                                 const float* __restrict__ b,
                                                 unsigned short* __restrict__ y,
                                                 int C) {
  const long long row = blockIdx.x;
  const int tid = threadIdx.x;
  const float* xr = x + row * C;
  float s = 0.0f, s2 = 0.0f;
  for (int c = tid; c < C; c += 256) { const float v = xr[c]; s += v; s2 += v * v; }
  __shared__ float r1[256], r2[256];
  r1[tid] = s; r2[tid] = s2; __syncthreads();
  for (int w = 128; w > 0; w >>= 1) {
    if (tid < w) { r1[tid] += r1[tid + w]; r2[tid] += r2[tid + w]; }
    __syncthreads();
  }
  const float mean = r1[0] / C;
  const float var  = r2[0] / C - mean * mean;
  const float rstd = rsqrtf(var + 1e-5f);
  for (int c = tid; c < C; c += 256)
    y[row * C + c] = f2bf((xr[c] - mean) * rstd * g[c] + b[c]);
}

// ---------------- causal softmax (scores fp32 -> probs bf16) ----------------
__global__ __launch_bounds__(256) void softmax_kernel(const float* __restrict__ sc,
                                                      unsigned short* __restrict__ pr) {
  const long long row = blockIdx.x;            // over B*H*S
  const int q = (int)(row % S_);
  const int tid = threadIdx.x;                 // = column
  const bool ok = tid <= q;
  float v = ok ? sc[row * S_ + tid] * 0.125f : -INFINITY;
  __shared__ float red[256];
  red[tid] = v; __syncthreads();
  for (int w = 128; w > 0; w >>= 1) {
    if (tid < w) red[tid] = fmaxf(red[tid], red[tid + w]);
    __syncthreads();
  }
  const float mx = red[0]; __syncthreads();
  const float e = ok ? __expf(v - mx) : 0.0f;
  red[tid] = e; __syncthreads();
  for (int w = 128; w > 0; w >>= 1) {
    if (tid < w) red[tid] += red[tid + w];
    __syncthreads();
  }
  pr[row * S_ + tid] = f2bf(e / red[0]);
}

// ---------------- router: noisy top-2 gating ----------------
__global__ __launch_bounds__(256) void router_kernel(const unsigned short* __restrict__ fx,
                                                     const float* __restrict__ rw,
                                                     const float* __restrict__ rb,
                                                     const float* __restrict__ nw,
                                                     const float* __restrict__ nb,
                                                     const float* __restrict__ noise,
                                                     float* __restrict__ gate) {
  const long long t = blockIdx.x;
  const int tid = threadIdx.x;
  float aL[E_], aN[E_];
#pragma unroll
  for (int e = 0; e < E_; ++e) { aL[e] = 0.0f; aN[e] = 0.0f; }
  for (int c = tid; c < O_; c += 256) {
    const float fv = bf2f(fx[t * O_ + c]);
    const float* rwc = rw + (long long)c * E_;
    const float* nwc = nw + (long long)c * E_;
#pragma unroll
    for (int e = 0; e < E_; ++e) { aL[e] += fv * rwc[e]; aN[e] += fv * nwc[e]; }
  }
  __shared__ float red[256];
  __shared__ float resL[E_], resN[E_];
  for (int e = 0; e < E_; ++e) {
    red[tid] = aL[e]; __syncthreads();
    for (int w = 128; w > 0; w >>= 1) { if (tid < w) red[tid] += red[tid + w]; __syncthreads(); }
    if (tid == 0) resL[e] = red[0];
    __syncthreads();
    red[tid] = aN[e]; __syncthreads();
    for (int w = 128; w > 0; w >>= 1) { if (tid < w) red[tid] += red[tid + w]; __syncthreads(); }
    if (tid == 0) resN[e] = red[0];
    __syncthreads();
  }
  if (tid == 0) {
    float noisy[E_];
#pragma unroll
    for (int e = 0; e < E_; ++e) {
      const float lg = resL[e] + rb[e];
      const float nl = resN[e] + nb[e];
      const float sp = (nl > 20.0f) ? nl : log1pf(__expf(nl));
      noisy[e] = lg + noise[t * E_ + e] * sp;
    }
    int i1 = 0; float v1 = noisy[0];
#pragma unroll
    for (int e = 1; e < E_; ++e) if (noisy[e] > v1) { v1 = noisy[e]; i1 = e; }
    int i2 = -1; float v2 = -INFINITY;
#pragma unroll
    for (int e = 0; e < E_; ++e) if (e != i1 && noisy[e] > v2) { v2 = noisy[e]; i2 = e; }
    const float e2 = __expf(v2 - v1);
    const float g1 = 1.0f / (1.0f + e2);
    const float g2 = e2 / (1.0f + e2);
#pragma unroll
    for (int e = 0; e < E_; ++e)
      gate[t * E_ + e] = (e == i1) ? g1 : ((e == i2) ? g2 : 0.0f);
  }
}

// ---------------- misc small kernels ----------------
__global__ __launch_bounds__(256) void zero_kernel(float* p, long long n) {
  const long long i = (long long)blockIdx.x * 256 + threadIdx.x;
  if (i < n) p[i] = 0.0f;
}

__global__ __launch_bounds__(256) void mean_kernel(const float* __restrict__ second,
                                                   float* __restrict__ tv) {
  const long long i = (long long)blockIdx.x * 256 + threadIdx.x;  // over B*O
  if (i >= (long long)B_ * O_) return;
  const long long b = i / O_;
  const int o = (int)(i % O_);
  float s = 0.0f;
  for (int ss = 0; ss < S_; ++ss) s += second[(b * S_ + ss) * (long long)O_ + o];
  tv[i] = s * (1.0f / S_);
}

__global__ __launch_bounds__(256) void cls_kernel(const float* __restrict__ tv,
                                                  const float* __restrict__ cw,
                                                  const float* __restrict__ cb,
                                                  float* __restrict__ out) {
  const int b = blockIdx.x;
  const int tid = threadIdx.x;
  float s = 0.0f;
  for (int c = tid; c < O_; c += 256) s += tv[(long long)b * O_ + c] * cw[c];
  __shared__ float red[256];
  red[tid] = s; __syncthreads();
  for (int w = 128; w > 0; w >>= 1) { if (tid < w) red[tid] += red[tid + w]; __syncthreads(); }
  if (tid == 0) out[b] = red[0] + cb[0];
}

// ---------------- host-side orchestration ----------------
static void run_gemm(hipStream_t stream, const GemmParams& p, int batches) {
  dim3 grid((p.M + BM - 1) / BM, (p.N + BN - 1) / BN, batches);
  hipLaunchKernelGGL(gemm_bf16_wmma, grid, dim3(256), 0, stream, p);
}

extern "C" void kernel_launch(void* const* d_in, const int* in_sizes, int n_in,
                              void* d_out, int out_size, void* d_ws, size_t ws_size,
                              hipStream_t stream) {
  (void)in_sizes; (void)n_in; (void)out_size; (void)ws_size;

  const int*   ids    = (const int*)d_in[0];
  const float* emb    = (const float*)d_in[2];
  const float* pe     = (const float*)d_in[3];
  const float* wq     = (const float*)d_in[4];
  const float* wk     = (const float*)d_in[5];
  const float* wv     = (const float*)d_in[6];
  const float* wo     = (const float*)d_in[7];
  const float* bo     = (const float*)d_in[8];
  const float* ln1_g  = (const float*)d_in[9];
  const float* ln1_b  = (const float*)d_in[10];
  const float* proj_w = (const float*)d_in[11];
  const float* proj_b = (const float*)d_in[12];
  const float* ln2_g  = (const float*)d_in[13];
  const float* ln2_b  = (const float*)d_in[14];
  const float* ln3_g  = (const float*)d_in[15];
  const float* ln3_b  = (const float*)d_in[16];
  const float* cls_w  = (const float*)d_in[17];
  const float* cls_b  = (const float*)d_in[18];

  // per-MoE-layer inputs: base 19 + 9*p
  const float* rW[2], *rB[2], *nW[2], *nB[2], *eW1[2], *eB1[2], *eW2[2], *eB2[2], *nz[2];
  for (int pl = 0; pl < 2; ++pl) {
    const int base = 19 + 9 * pl;
    rW[pl]  = (const float*)d_in[base + 0];
    rB[pl]  = (const float*)d_in[base + 1];
    nW[pl]  = (const float*)d_in[base + 2];
    nB[pl]  = (const float*)d_in[base + 3];
    eW1[pl] = (const float*)d_in[base + 4];
    eB1[pl] = (const float*)d_in[base + 5];
    eW2[pl] = (const float*)d_in[base + 6];
    eB2[pl] = (const float*)d_in[base + 7];
    nz[pl]  = (const float*)d_in[base + 8];
  }

  // ---- workspace carve (256B aligned) ----
  char* ws = (char*)d_ws;
  size_t off = 0;
  auto alloc = [&](size_t bytes) -> char* {
    char* p = ws + off;
    off += (bytes + 255) & ~(size_t)255;
    return p;
  };
  float*          x0     = (float*)alloc((size_t)T_ * D_ * 4);
  float*          scores = (float*)alloc((size_t)B_ * H_ * S_ * S_ * 4); // aliased by Hbuf later
  unsigned short* Hbuf   = (unsigned short*)scores;                      // T_*FF_*2 == same size
  unsigned short* probs  = (unsigned short*)alloc((size_t)B_ * H_ * S_ * S_ * 2);
  unsigned short* ln1b   = (unsigned short*)alloc((size_t)T_ * D_ * 2);
  unsigned short* qb     = (unsigned short*)alloc((size_t)T_ * D_ * 2);
  unsigned short* kb     = (unsigned short*)alloc((size_t)T_ * D_ * 2);
  unsigned short* vb     = (unsigned short*)alloc((size_t)T_ * D_ * 2);
  unsigned short* attnb  = (unsigned short*)alloc((size_t)T_ * D_ * 2);
  unsigned short* x1b    = (unsigned short*)alloc((size_t)T_ * D_ * 2);
  float*          xo     = (float*)alloc((size_t)T_ * O_ * 4);
  unsigned short* ln2b   = (unsigned short*)alloc((size_t)T_ * O_ * 2);
  unsigned short* ln3b   = (unsigned short*)alloc((size_t)T_ * O_ * 2);
  float*          gate1  = (float*)alloc((size_t)T_ * E_ * 4);
  float*          gate2  = (float*)alloc((size_t)T_ * E_ * 4);

  float* outF = (float*)d_out;                 // first  [T_,O_]
  float* outS = outF + (size_t)T_ * O_;        // second [T_,O_]
  float* tv   = outS + (size_t)T_ * O_;        // [B_,O_]
  float* cls  = tv + (size_t)B_ * O_;          // [B_]

  const int F_RELU = 1, F_BF32 = 2, F_BT = 4, F_ST16 = 8, F_ACC = 16;

  // 1) embed + pe
  hipLaunchKernelGGL(embed_kernel, dim3((T_ * D_) / 256), dim3(256), 0, stream,
                     ids, emb, pe, x0);
  // 2) ln1
  hipLaunchKernelGGL(ln_kernel, dim3(T_), dim3(256), 0, stream, x0, ln1_g, ln1_b, ln1b, D_);

  // 3) q, k, v = ln1 @ w{q,k,v}  -> bf16
  {
    GemmParams p{}; p.A = ln1b; p.lda = D_; p.M = T_; p.N = D_; p.Kd = D_;
    p.ldb = D_; p.ldc = D_; p.Hn = 1; p.flags = F_BF32 | F_ST16;
    p.B = wq; p.Cb = qb; run_gemm(stream, p, 1);
    p.B = wk; p.Cb = kb; run_gemm(stream, p, 1);
    p.B = wv; p.Cb = vb; run_gemm(stream, p, 1);
  }
  // 4) scores[b,h] = q[b,:,h,:] @ k[b,:,h,:]^T   (B pre-transposed mode)
  {
    GemmParams p{}; p.A = qb; p.lda = D_; p.sAb = (long long)S_ * D_; p.sAh = HS_;
    p.B = kb; p.ldb = D_; p.sBb = (long long)S_ * D_; p.sBh = HS_;
    p.Cf = scores; p.ldc = S_; p.sCb = (long long)H_ * S_ * S_; p.sCh = (long long)S_ * S_;
    p.M = S_; p.N = S_; p.Kd = HS_; p.Hn = H_; p.flags = F_BT;
    run_gemm(stream, p, B_ * H_);
  }
  // 5) causal softmax (* HS^-0.5)
  hipLaunchKernelGGL(softmax_kernel, dim3(B_ * H_ * S_), dim3(256), 0, stream, scores, probs);
  // 6) attn[b,h] = probs @ v[b,:,h,:] -> bf16 into [T_,D_] at head offset
  {
    GemmParams p{}; p.A = probs; p.lda = S_;
    p.sAb = (long long)H_ * S_ * S_; p.sAh = (long long)S_ * S_;
    p.B = vb; p.ldb = D_; p.sBb = (long long)S_ * D_; p.sBh = HS_;
    p.Cb = attnb; p.ldc = D_; p.sCb = (long long)S_ * D_; p.sCh = HS_;
    p.M = S_; p.N = HS_; p.Kd = S_; p.Hn = H_; p.flags = F_ST16;
    run_gemm(stream, p, B_ * H_);
  }
  // 7) x1 = x0 + attn @ wo + bo  -> bf16
  {
    GemmParams p{}; p.A = attnb; p.lda = D_; p.B = wo; p.ldb = D_; p.bias = bo;
    p.addRes = x0; p.resLd = D_; p.Cb = x1b; p.ldc = D_;
    p.M = T_; p.N = D_; p.Kd = D_; p.Hn = 1; p.flags = F_BF32 | F_ST16;
    run_gemm(stream, p, 1);
  }
  // 8) xo = x1 @ proj_w + proj_b  -> fp32
  {
    GemmParams p{}; p.A = x1b; p.lda = D_; p.B = proj_w; p.ldb = O_; p.bias = proj_b;
    p.Cf = xo; p.ldc = O_; p.M = T_; p.N = O_; p.Kd = D_; p.Hn = 1; p.flags = F_BF32;
    run_gemm(stream, p, 1);
  }
  // 9) ln2 / ln3
  hipLaunchKernelGGL(ln_kernel, dim3(T_), dim3(256), 0, stream, xo, ln2_g, ln2_b, ln2b, O_);
  hipLaunchKernelGGL(ln_kernel, dim3(T_), dim3(256), 0, stream, xo, ln3_g, ln3_b, ln3b, O_);
  // 10) routing
  hipLaunchKernelGGL(router_kernel, dim3(T_), dim3(256), 0, stream,
                     ln2b, rW[0], rB[0], nW[0], nB[0], nz[0], gate1);
  hipLaunchKernelGGL(router_kernel, dim3(T_), dim3(256), 0, stream,
                     ln3b, rW[1], rB[1], nW[1], nB[1], nz[1], gate2);
  // 11) zero first+second output regions (accumulated into)
  hipLaunchKernelGGL(zero_kernel, dim3((2 * T_ * O_) / 256), dim3(256), 0, stream,
                     outF, (long long)2 * T_ * O_);

  // 12) MoE layers: for each expert  H = relu(fx@w1+b1);  out += gate[:,e]*(H@w2+b2)
  for (int pl = 0; pl < 2; ++pl) {
    const unsigned short* fx = pl ? ln3b : ln2b;
    const float* gate = pl ? gate2 : gate1;
    float* outp = pl ? outS : outF;
    for (int e = 0; e < E_; ++e) {
      GemmParams g1{}; g1.A = fx; g1.lda = O_;
      g1.B = eW1[pl] + (long long)e * O_ * FF_; g1.ldb = FF_;
      g1.bias = eB1[pl] + (long long)e * FF_;
      g1.Cb = Hbuf; g1.ldc = FF_;
      g1.M = T_; g1.N = FF_; g1.Kd = O_; g1.Hn = 1;
      g1.flags = F_BF32 | F_ST16 | F_RELU;
      run_gemm(stream, g1, 1);

      GemmParams g2{}; g2.A = Hbuf; g2.lda = FF_;
      g2.B = eW2[pl] + (long long)e * FF_ * O_; g2.ldb = O_;
      g2.bias = eB2[pl] + (long long)e * O_;
      g2.Cf = outp; g2.ldc = O_;
      g2.rowScale = gate + e; g2.rsStride = E_;
      g2.M = T_; g2.N = O_; g2.Kd = FF_; g2.Hn = 1;
      g2.flags = F_BF32 | F_ACC;
      run_gemm(stream, g2, 1);
    }
  }

  // 13) text_vector = mean_s(second);  text_cls = tv @ cls_w + cls_b
  hipLaunchKernelGGL(mean_kernel, dim3((B_ * O_ + 255) / 256), dim3(256), 0, stream, outS, tv);
  hipLaunchKernelGGL(cls_kernel, dim3(B_), dim3(256), 0, stream, tv, cls_w, cls_b, cls);
}
